// TransformerForTranslation_62972810494670
// MI455X (gfx1250) — compile-verified, matrix operand order checked by reading, not run on hardware
//
#include <hip/hip_runtime.h>
#include <hip/hip_bf16.h>
#include <cstddef>
#include <cstdint>

// ---------------------------------------------------------------------------
// Types for CDNA5 WMMA (wave32, 16x16x32 bf16 -> f32)
// ---------------------------------------------------------------------------
typedef __bf16 bf16_t;
typedef __attribute__((ext_vector_type(4)))  bf16_t v4bf;
typedef __attribute__((ext_vector_type(8)))  bf16_t v8bf;
typedef __attribute__((ext_vector_type(16))) bf16_t v16bf;
typedef __attribute__((ext_vector_type(8)))  float  v8f;

static __device__ inline v8f vzero8() {
  v8f z;
#pragma unroll
  for (int i = 0; i < 8; ++i) z[i] = 0.0f;
  return z;
}

static __device__ inline v8f wmma_bf16(v16bf a, v16bf b, v8f c) {
  // D = A(16x32 bf16) * B(32x16 bf16) + C(16x16 f32)
  return __builtin_amdgcn_wmma_f32_16x16x32_bf16(
      /*neg_a=*/false, a, /*neg_b=*/false, b,
      /*c_mod=*/(short)0, c, /*reuse_a=*/false, /*reuse_b=*/false);
}

// Async 16-byte global->LDS copy (CDNA5 VGLOBAL async path, ASYNCcnt).
// lds_addr: byte offset within the workgroup's LDS allocation (low 32 bits of
// a generic pointer to __shared__); gaddr: 64-bit global byte address.
static __device__ inline void async_copy16(unsigned lds_addr,
                                           unsigned long long gaddr) {
  asm volatile("global_load_async_to_lds_b128 %0, %1, off"
               :: "v"(lds_addr), "v"(gaddr) : "memory");
}
static __device__ inline void wait_async0() {
  asm volatile("s_wait_asynccnt 0x0" ::: "memory");
}

// Load a 16x32 bf16 fragment from an LDS tile stored row-major with stride
// `ld` (elements). Works for A (rows = M) and for B staged n-major (rows = N).
// ISA layout (16-bit, 16x32): lanes 0-15 -> row = lane, K = {0..7, 16..23};
// lanes 16-31 -> row = lane-16, K = {8..15, 24..31}.
static __device__ inline v16bf load_frag(const bf16_t* base, int ld) {
  const int lane = threadIdx.x & 31;
  const int row  = lane & 15;
  const int kb   = (lane >> 4) << 3;  // 0 or 8
  const bf16_t* p = base + row * ld;
  v8bf lo = *(const v8bf*)(p + kb);        // K = kb .. kb+7
  v8bf hi = *(const v8bf*)(p + kb + 16);   // K = kb+16 .. kb+23
  v16bf out;
#pragma unroll
  for (int i = 0; i < 8; ++i) { out[i] = lo[i]; out[i + 8] = hi[i]; }
  return out;
}

// ---------------------------------------------------------------------------
// Weight transpose + convert: Wt[n][k] = (bf16)W[k][n]. 32x32 LDS tiles,
// coalesced on both sides. Grid: (N/32, K/32), 256 threads.
// ---------------------------------------------------------------------------
__global__ void __launch_bounds__(256)
transpose_cvt_kernel(const float* __restrict__ W, bf16_t* __restrict__ Wt,
                     int K, int N) {
  __shared__ float tile[32][33];
  const int tx = threadIdx.x & 31;
  const int ty = threadIdx.x >> 5;  // 0..7
  const int n0 = blockIdx.x * 32;
  const int k0 = blockIdx.y * 32;
#pragma unroll
  for (int p = 0; p < 4; ++p)
    tile[ty + p * 8][tx] = W[(size_t)(k0 + ty + p * 8) * N + n0 + tx];
  __syncthreads();
#pragma unroll
  for (int p = 0; p < 4; ++p)
    Wt[(size_t)(n0 + ty + p * 8) * K + k0 + tx] = (bf16_t)tile[tx][ty + p * 8];
}

// ---------------------------------------------------------------------------
// GEMM: C[M,N] = act(A[M,K] @ W[K,N] + bias[N]). A is bf16 row-major,
// B is bf16 n-major (Wt[N][K]). Tiles staged via async global->LDS copies,
// double-buffered (copies for step t+1 overlap WMMA of step t).
// Block: 256 threads (8 waves), tile 128x128, K-step 32.
// Optional f32 and/or bf16 outputs (either may be null).
// ---------------------------------------------------------------------------
#define BM 128
#define BN 128
#define BK 32

__global__ void __launch_bounds__(256)
gemm_async_kernel(const bf16_t* __restrict__ A, const bf16_t* __restrict__ Bt,
                  const float* __restrict__ bias, float* __restrict__ Cf,
                  bf16_t* __restrict__ Cb, int M, int N, int K, int relu) {
  __shared__ bf16_t sA[2][BM * BK];  // [m][k], 8 KB each
  __shared__ bf16_t sB[2][BN * BK];  // [n][k], 8 KB each

  const int tid  = threadIdx.x;
  const int lane = tid & 31;
  const int wid  = tid >> 5;
  const int wm   = wid >> 2;   // 0..1
  const int wn   = wid & 3;    // 0..3
  const int m0   = blockIdx.y * BM;
  const int n0   = blockIdx.x * BN;

  const unsigned ldsA0 = (unsigned)(uintptr_t)(&sA[0][0]);
  const unsigned ldsB0 = (unsigned)(uintptr_t)(&sB[0][0]);
  const unsigned long long gA = (unsigned long long)(uintptr_t)A;
  const unsigned long long gB = (unsigned long long)(uintptr_t)Bt;

  // Issue async copies for one 128x32 bf16 tile pair (A and B).
  // Per tile: 128 rows x 64 bytes = 512 x 16B chunks = 2 issues/thread.
  auto issue_tiles = [&](int buf, int kt) {
#pragma unroll
    for (int i = 0; i < 2; ++i) {
      const int cid = tid + i * 256;
      const int row = cid >> 2;          // 0..127
      const int cb  = (cid & 3) << 4;    // byte offset in 64B row
      async_copy16(ldsA0 + (unsigned)(buf * BM * BK * 2 + row * 64 + cb),
                   gA + (((size_t)(m0 + row) * K + kt) << 1) + cb);
      async_copy16(ldsB0 + (unsigned)(buf * BN * BK * 2 + row * 64 + cb),
                   gB + (((size_t)(n0 + row) * K + kt) << 1) + cb);
    }
  };

  v8f acc[4][2];
#pragma unroll
  for (int i = 0; i < 4; ++i)
#pragma unroll
    for (int j = 0; j < 2; ++j) acc[i][j] = vzero8();

  issue_tiles(0, 0);
  int cur = 0;
  for (int kt = 0; kt < K; kt += BK) {
    wait_async0();        // this wave's copies into sX[cur] have landed
    __syncthreads();      // everyone's copies have landed; prev compute done
    if (kt + BK < K) issue_tiles(cur ^ 1, kt + BK);  // overlap with WMMA

    v16bf afrag[4], bfrag[2];
#pragma unroll
    for (int i = 0; i < 4; ++i)
      afrag[i] = load_frag(&sA[cur][(wm * 64 + i * 16) * BK], BK);
#pragma unroll
    for (int j = 0; j < 2; ++j)
      bfrag[j] = load_frag(&sB[cur][(wn * 32 + j * 16) * BK], BK);
#pragma unroll
    for (int i = 0; i < 4; ++i)
#pragma unroll
      for (int j = 0; j < 2; ++j)
        acc[i][j] = wmma_bf16(afrag[i], bfrag[j], acc[i][j]);
    cur ^= 1;
  }

  // Epilogue. C layout: VGPR r -> M = r + 8*(lane>=16), N = lane&15.
  const int col  = lane & 15;
  const int rsel = (lane >> 4) << 3;
#pragma unroll
  for (int i = 0; i < 4; ++i) {
#pragma unroll
    for (int j = 0; j < 2; ++j) {
      const int gm = m0 + wm * 64 + i * 16;
      const int gn = n0 + wn * 32 + j * 16;
      const float bv = bias[gn + col];
#pragma unroll
      for (int r = 0; r < 8; ++r) {
        float val = acc[i][j][r] + bv;
        if (relu) val = fmaxf(val, 0.0f);
        const size_t idx = (size_t)(gm + rsel + r) * N + gn + col;
        if (Cf) Cf[idx] = val;
        if (Cb) Cb[idx] = (bf16_t)val;
      }
    }
  }
}

// ---------------------------------------------------------------------------
// Attention: one block per (b, h, 64-query tile). S=256, dk=64, D=512 fixed.
// Inputs Q/K/V are bf16 [B,S,D] (head h at column h*64). Output bf16.
// Q (64x64) and K (256x64) staged with async copies; S = QK^T/8 via WMMA;
// masked softmax (P as bf16); V^T staged via LDS scatter; O = P V via WMMA.
// causal: also require k > q (reference's non-inverted look-ahead mask).
// ---------------------------------------------------------------------------
__global__ void __launch_bounds__(256)
attn_kernel(const bf16_t* __restrict__ Qg, const bf16_t* __restrict__ Kg,
            const bf16_t* __restrict__ Vg, bf16_t* __restrict__ Og,
            const int* __restrict__ toks, int causal) {
  extern __shared__ char smem[];
  bf16_t* sQ = (bf16_t*)(smem);                          //  64x64  =  8 KB
  bf16_t* sK = (bf16_t*)(smem + 8192);                   // 256x64  = 32 KB
  float*  sS = (float*) (smem + 8192 + 32768);           //  64x256 = 64 KB
  bf16_t* sP = (bf16_t*)(smem + 8192 + 32768 + 65536);   //  64x256 = 32 KB

  const int tid  = threadIdx.x;
  const int lane = tid & 31;
  const int w    = tid >> 5;       // wave 0..7
  const int bh   = blockIdx.y;
  const int b    = bh >> 3;
  const int h    = bh & 7;
  const int q0   = blockIdx.x * 64;
  const size_t base = (size_t)b * 256 * 512 + (size_t)h * 64;  // elements

  const unsigned ldsQ = (unsigned)(uintptr_t)sQ;
  const unsigned ldsK = (unsigned)(uintptr_t)sK;
  const unsigned long long gQ = (unsigned long long)(uintptr_t)Qg;
  const unsigned long long gK = (unsigned long long)(uintptr_t)Kg;

  // Phase 1: async-stage Q (512 chunks) and K (2048 chunks); 16B per chunk,
  // each row's head slice is 64 bf16 = 128B = 8 chunks.
#pragma unroll
  for (int i = 0; i < 2; ++i) {
    const int cid = tid + i * 256;
    const int row = cid >> 3, c = (cid & 7) << 4;
    async_copy16(ldsQ + (unsigned)(row * 128 + c),
                 gQ + ((base + (size_t)(q0 + row) * 512) << 1) + c);
  }
#pragma unroll
  for (int i = 0; i < 8; ++i) {
    const int cid = tid + i * 256;
    const int row = cid >> 3, c = (cid & 7) << 4;
    async_copy16(ldsK + (unsigned)(row * 128 + c),
                 gK + ((base + (size_t)row * 512) << 1) + c);
  }
  wait_async0();
  __syncthreads();

  // Phase 2: scores. Wave w -> q-subtile (w>>1), key half (w&1)*128.
  const int qs    = (w >> 1) * 16;
  const int khalf = (w & 1) * 128;
  const int col   = lane & 15;
  const int rsel  = (lane >> 4) << 3;
#pragma unroll
  for (int ks = 0; ks < 8; ++ks) {
    const int kb = khalf + ks * 16;
    v8f acc = vzero8();
#pragma unroll
    for (int kk = 0; kk < 64; kk += 32) {
      v16bf a  = load_frag(sQ + qs * 64 + kk, 64);
      v16bf bb = load_frag(sK + kb * 64 + kk, 64);
      acc = wmma_bf16(a, bb, acc);
    }
#pragma unroll
    for (int r = 0; r < 8; ++r)
      sS[(qs + rsel + r) * 256 + kb + col] = acc[r] * 0.125f;  // 1/sqrt(64)
  }
  __syncthreads();

  // Phase 3: mask + softmax; wave w handles rows w*8 .. w*8+7.
#pragma unroll 1
  for (int rr = 0; rr < 8; ++rr) {
    const int row  = w * 8 + rr;
    const int qpos = q0 + row;
    float vals[8];
    float mx = -3.0e38f;
#pragma unroll
    for (int i = 0; i < 8; ++i) {
      const int c = lane + i * 32;
      float s = sS[row * 256 + c];
      bool keep = (toks[b * 256 + c] != 0);
      if (causal) keep = keep && (c > qpos);
      s = keep ? s : -1.0e9f;
      vals[i] = s;
      mx = fmaxf(mx, s);
    }
#pragma unroll
    for (int off = 16; off > 0; off >>= 1)
      mx = fmaxf(mx, __shfl_xor(mx, off, 32));
    float sum = 0.0f;
#pragma unroll
    for (int i = 0; i < 8; ++i) { vals[i] = __expf(vals[i] - mx); sum += vals[i]; }
#pragma unroll
    for (int off = 16; off > 0; off >>= 1)
      sum += __shfl_xor(sum, off, 32);
    const float inv = 1.0f / sum;
#pragma unroll
    for (int i = 0; i < 8; ++i)
      sP[row * 256 + lane + i * 32] = (bf16_t)(vals[i] * inv);
  }
  __syncthreads();

  // Phase 4: stage V transposed (sVt[dk][key]) over sK's region.
  bf16_t* sVt = sK;
#pragma unroll
  for (int i = 0; i < 16; ++i) {
    const int v = tid + i * 256;
    const int key = v >> 4;
    const int c4  = (v & 15) << 2;
    v4bf d = *(const v4bf*)(Vg + base + (size_t)key * 512 + c4);
    sVt[(c4 + 0) * 256 + key] = d[0];
    sVt[(c4 + 1) * 256 + key] = d[1];
    sVt[(c4 + 2) * 256 + key] = d[2];
    sVt[(c4 + 3) * 256 + key] = d[3];
  }
  __syncthreads();

  // Phase 5: O = P @ V. Wave w -> q-subtile (w>>1), dk tile pair (w&1)*2.
  const int npair = (w & 1) * 2;
  v8f oacc[2] = {vzero8(), vzero8()};
#pragma unroll
  for (int kk = 0; kk < 256; kk += 32) {
    v16bf a = load_frag(sP + qs * 256 + kk, 256);
#pragma unroll
    for (int j = 0; j < 2; ++j) {
      v16bf bb = load_frag(sVt + (npair + j) * 16 * 256 + kk, 256);
      oacc[j] = wmma_bf16(a, bb, oacc[j]);
    }
  }
#pragma unroll
  for (int j = 0; j < 2; ++j)
#pragma unroll
    for (int r = 0; r < 8; ++r)
      Og[base + (size_t)(q0 + qs + rsel + r) * 512 + (npair + j) * 16 + col] =
          (bf16_t)oacc[j][r];
}

// ---------------------------------------------------------------------------
// Residual + LayerNorm: out = g*(x+t - mean)/(std_unbiased + eps) + b.
// One block per row (512 cols); writes f32 stream + bf16 shadow for GEMMs.
// ---------------------------------------------------------------------------
__global__ void __launch_bounds__(256)
ln_res_kernel(const float* __restrict__ x, const float* __restrict__ t,
              const float* __restrict__ g, const float* __restrict__ bb,
              float* __restrict__ outf, bf16_t* __restrict__ outb) {
  __shared__ float red[256];
  const int row = blockIdx.x;
  const int tid = threadIdx.x;
  const size_t ro = (size_t)row * 512;

  const float v0 = x[ro + tid] + t[ro + tid];
  const float v1 = x[ro + tid + 256] + t[ro + tid + 256];

  red[tid] = v0 + v1;
  __syncthreads();
  for (int st = 128; st > 0; st >>= 1) {
    if (tid < st) red[tid] += red[tid + st];
    __syncthreads();
  }
  const float mean = red[0] * (1.0f / 512.0f);
  __syncthreads();

  const float d0 = v0 - mean, d1 = v1 - mean;
  red[tid] = d0 * d0 + d1 * d1;
  __syncthreads();
  for (int st = 128; st > 0; st >>= 1) {
    if (tid < st) red[tid] += red[tid + st];
    __syncthreads();
  }
  const float var = red[0] * (1.0f / 511.0f);  // unbiased (ddof=1)
  const float inv = 1.0f / (sqrtf(var) + 1e-6f);

  const float o0 = g[tid] * d0 * inv + bb[tid];
  const float o1 = g[tid + 256] * d1 * inv + bb[tid + 256];
  outf[ro + tid]       = o0;
  outf[ro + tid + 256] = o1;
  outb[ro + tid]       = (bf16_t)o0;
  outb[ro + tid + 256] = (bf16_t)o1;
}

// ---------------------------------------------------------------------------
// Embedding + sinusoidal positional encoding (f32 + bf16 shadow outputs).
// ---------------------------------------------------------------------------
__global__ void __launch_bounds__(256)
embed_kernel(const int* __restrict__ toks, const float* __restrict__ emb,
             float* __restrict__ outf, bf16_t* __restrict__ outb) {
  const int row = blockIdx.x;     // b*256 + s
  const int s   = row & 255;      // position in sequence (S = 256)
  const int tid = threadIdx.x;
  const int tok = toks[row];
#pragma unroll
  for (int half = 0; half < 2; ++half) {
    const int col = tid + half * 256;
    const int i2  = (col >> 1) << 1;  // 2i
    const float div = __expf(-9.210340371976184f * (float)i2 * (1.0f / 512.0f));
    const float ang = (float)s * div;
    const float pe  = (col & 1) ? __cosf(ang) : __sinf(ang);
    const float v = emb[(size_t)tok * 512 + col] + pe;
    outf[(size_t)row * 512 + col] = v;
    outb[(size_t)row * 512 + col] = (bf16_t)v;
  }
}

// ---------------------------------------------------------------------------
// Host orchestration.
// Input leaf order (setup_inputs() insertion order):
//   0:src 1:tgt 2:src_emb 3:tgt_emb
//   enc layer l (base 4+16l): Wq bq Wk bk Wv bv Wo bo W1 b1 W2 b2 g1 be1 g2 be2
//   dec layer l (base 100+26l): sWq..sbo(0-7) cWq..cbo(8-15) W1 b1 W2 b2
//                               g1 be1 g2 be2 g3 be3 (16-25)
//   256: out_W  257: out_b
// ---------------------------------------------------------------------------
extern "C" void kernel_launch(void* const* d_in, const int* in_sizes, int n_in,
                              void* d_out, int out_size, void* d_ws,
                              size_t ws_size, hipStream_t stream) {
  (void)in_sizes; (void)n_in; (void)out_size; (void)ws_size;

  const int*   src     = (const int*)d_in[0];
  const int*   tgt     = (const int*)d_in[1];
  const float* src_emb = (const float*)d_in[2];
  const float* tgt_emb = (const float*)d_in[3];
  auto P = [&](int i) { return (const float*)d_in[i]; };

  const size_t ACT = (size_t)4096 * 512;  // B*S x D elements
  float* ws = (float*)d_ws;
  float* X = ws;            // encoder stream f32 (becomes `mem`)
  float* Y = X + ACT;       // decoder stream f32
  float* T = Y + ACT;       // sublayer output f32 (pre-LN)
  bf16_t* bws = (bf16_t*)(T + ACT);
  bf16_t* Xbf = bws;                 // bf16 shadows feeding GEMMs
  bf16_t* Ybf = Xbf + ACT;
  bf16_t* Qbf = Ybf + ACT;
  bf16_t* Kbf = Qbf + ACT;
  bf16_t* Vbf = Kbf + ACT;
  bf16_t* Obf = Vbf + ACT;
  bf16_t* Hbf = Obf + ACT;                       // 4096 x 2048
  bf16_t* Wt  = Hbf + (size_t)4096 * 2048;       // up to 32000 x 512

  const dim3 blk(256);
  // transpose-convert weight, then async WMMA GEMM
  auto gemm = [&](const bf16_t* A, const float* Wf, const float* bias,
                  float* Cf, bf16_t* Cb, int M, int N, int K, int relu) {
    dim3 tg(N / 32, K / 32);
    transpose_cvt_kernel<<<tg, blk, 0, stream>>>(Wf, Wt, K, N);
    dim3 grid(N / BN, M / BM);
    gemm_async_kernel<<<grid, blk, 0, stream>>>(A, Wt, bias, Cf, Cb, M, N, K,
                                                relu);
  };
  const size_t attn_smem = 8192 + 32768 + 65536 + 32768;  // 136 KB
  auto attn = [&](const bf16_t* Q, const bf16_t* K, const bf16_t* V, bf16_t* O,
                  const int* toks, int causal) {
    dim3 grid(4, 128);  // S/64 query tiles, B*H
    attn_kernel<<<grid, blk, attn_smem, stream>>>(Q, K, V, O, toks, causal);
  };
  auto lnres = [&](const float* x, const float* t, const float* g,
                   const float* b, float* of, bf16_t* ob) {
    ln_res_kernel<<<dim3(4096), blk, 0, stream>>>(x, t, g, b, of, ob);
  };

  // ---- Encoder ----
  embed_kernel<<<dim3(4096), blk, 0, stream>>>(src, src_emb, X, Xbf);
  for (int l = 0; l < 6; ++l) {
    const int p = 4 + l * 16;
    gemm(Xbf, P(p + 0), P(p + 1), nullptr, Qbf, 4096, 512, 512, 0);
    gemm(Xbf, P(p + 2), P(p + 3), nullptr, Kbf, 4096, 512, 512, 0);
    gemm(Xbf, P(p + 4), P(p + 5), nullptr, Vbf, 4096, 512, 512, 0);
    attn(Qbf, Kbf, Vbf, Obf, src, 0);
    gemm(Obf, P(p + 6), P(p + 7), T, nullptr, 4096, 512, 512, 0);
    lnres(X, T, P(p + 12), P(p + 13), X, Xbf);
    gemm(Xbf, P(p + 8), P(p + 9), nullptr, Hbf, 4096, 2048, 512, 1);  // ReLU
    gemm(Hbf, P(p + 10), P(p + 11), T, nullptr, 4096, 512, 2048, 0);
    lnres(X, T, P(p + 14), P(p + 15), X, Xbf);
  }
  // X/Xbf == mem

  // ---- Decoder ----
  embed_kernel<<<dim3(4096), blk, 0, stream>>>(tgt, tgt_emb, Y, Ybf);
  for (int l = 0; l < 6; ++l) {
    const int p = 100 + l * 26;
    // self-attention (reference's non-inverted look-ahead mask: k > q)
    gemm(Ybf, P(p + 0), P(p + 1), nullptr, Qbf, 4096, 512, 512, 0);
    gemm(Ybf, P(p + 2), P(p + 3), nullptr, Kbf, 4096, 512, 512, 0);
    gemm(Ybf, P(p + 4), P(p + 5), nullptr, Vbf, 4096, 512, 512, 0);
    attn(Qbf, Kbf, Vbf, Obf, tgt, 1);
    gemm(Obf, P(p + 6), P(p + 7), T, nullptr, 4096, 512, 512, 0);
    lnres(Y, T, P(p + 20), P(p + 21), Y, Ybf);
    // cross-attention (keys/values from encoder memory)
    gemm(Ybf, P(p + 8), P(p + 9), nullptr, Qbf, 4096, 512, 512, 0);
    gemm(Xbf, P(p + 10), P(p + 11), nullptr, Kbf, 4096, 512, 512, 0);
    gemm(Xbf, P(p + 12), P(p + 13), nullptr, Vbf, 4096, 512, 512, 0);
    attn(Qbf, Kbf, Vbf, Obf, src, 0);
    gemm(Obf, P(p + 14), P(p + 15), T, nullptr, 4096, 512, 512, 0);
    lnres(Y, T, P(p + 22), P(p + 23), Y, Ybf);
    // FFN
    gemm(Ybf, P(p + 16), P(p + 17), nullptr, Hbf, 4096, 2048, 512, 1);
    gemm(Hbf, P(p + 18), P(p + 19), T, nullptr, 4096, 512, 2048, 0);
    lnres(Y, T, P(p + 24), P(p + 25), Y, Ybf);
  }

  // ---- Output projection: [4096,512] @ [512,32000] + bias -> f32 logits ----
  gemm(Ybf, P(256), P(257), (float*)d_out, nullptr, 4096, 32000, 512, 0);
}
